// AutoCorrelation_55705725829700
// MI455X (gfx1250) — matrix operand order repaired and checked by reading, not compiled, not run on hardware
//
#include <hip/hip_runtime.h>
#include <hip/hip_bf16.h>
#include <math.h>

// ---------------------------------------------------------------------------
// AutoCorrelation block for MI455X (gfx1250, wave32, WMMA, async-to-LDS).
// All GEMM operands kept f16 in global memory so tile staging is pure
// GLOBAL_LOAD_ASYNC_TO_LDS_B128 DMA; f32 only for spectra/corr/fc/LN math.
// ---------------------------------------------------------------------------

typedef _Float16 f16;
typedef __attribute__((ext_vector_type(16))) _Float16 v16h;
typedef __attribute__((ext_vector_type(8)))  _Float16 v8h;
typedef __attribute__((ext_vector_type(8)))  float    v8f;
typedef int v4i __attribute__((vector_size(16)));   // matches builtin param type

typedef __attribute__((address_space(1))) v4i* gptr_b128;   // global int4*
typedef __attribute__((address_space(3))) v4i* lptr_b128;   // LDS int4*

#define BB      8
#define LL      2048
#define DM      512
#define NCH     4096        // B*H*DK channels
#define NF2     2176        // padded 2*(L/2+1)=2050 -> multiple of 128
#define TOPK    38          // int(5*ln(2048))
#define LN_EPS  1e-5f

// ----------------------- CDNA5 async-to-LDS helpers ------------------------

__device__ __forceinline__ void async_copy16(const f16* src, f16* dstLds)
{
#if __has_builtin(__builtin_amdgcn_global_load_async_to_lds_b128)
    __builtin_amdgcn_global_load_async_to_lds_b128(
        (gptr_b128)(uintptr_t)src,
        (lptr_b128)(uint32_t)(uintptr_t)dstLds,
        0, 0);
#else
    asm volatile("global_load_async_to_lds_b128 %0, %1, off"
                 :: "v"((uint32_t)(uintptr_t)dstLds), "v"(src)
                 : "memory");
#endif
}

__device__ __forceinline__ void wait_async0()
{
#if __has_builtin(__builtin_amdgcn_s_wait_asynccnt)
    __builtin_amdgcn_s_wait_asynccnt(0);
#else
    asm volatile("s_wait_asynccnt 0x0" ::: "memory");
#endif
}

// ---------------------------- WMMA GEMM ------------------------------------
// C[M,N] = A[M,K](f16) @ Bt[N,K](f16, pre-transposed), f32 accum.
// Block: 256 threads = 8 waves, 128x128 tile; wave = 64x32 (4x2 WMMA frags).
// Double-buffered LDS, tiles staged with async b128 copies.
// mode 0: f32 C row-major.  mode 1: f16 scatter channel-major
//   C[((b*512+n)*2048)+l], m = b*2048+l (N must be 512).  mode 2: f16 row-major.

#define BK  32
#define LP  40   // LDS pitch in halves: 80B rows keep 16B alignment

__global__ __launch_bounds__(256)
void gemm_wmma_f16(const f16* __restrict__ Ag, const f16* __restrict__ Bg,
                   void* __restrict__ Cm, int M, int N, int K, int mode)
{
    __shared__ f16 As[2][128][LP];
    __shared__ f16 Bs[2][128][LP];

    const int tid   = threadIdx.x;
    const int lane  = tid & 31;
    const int wid   = tid >> 5;            // 0..7
    const int wm    = (wid >> 2) * 64;     // 0 / 64
    const int wn    = (wid & 3) * 32;      // 0/32/64/96
    const int m0    = blockIdx.y * 128;
    const int n0    = blockIdx.x * 128;
    const int l16   = lane & 15;
    const int khalf = lane >> 4;           // 0/1

    v8f acc[4][2] = {};

    const int nk = K / BK;

    // prologue: stage tile 0 into buffer 0
#pragma unroll
    for (int p = 0; p < 2; ++p) {
        int chunk = p * 256 + tid;         // 0..511
        int r  = chunk >> 2;               // 0..127
        int cc = (chunk & 3) * 8;          // halves within BK
        async_copy16(Ag + (size_t)(m0 + r) * K + cc, &As[0][r][cc]);
        async_copy16(Bg + (size_t)(n0 + r) * K + cc, &Bs[0][r][cc]);
    }

    for (int kt = 0; kt < nk; ++kt) {
        const int cur = kt & 1;
        wait_async0();
        __syncthreads();                    // buffer `cur` ready for all waves

        if (kt + 1 < nk) {                  // stage next tile while computing
            const int nxt = cur ^ 1;
            const int k0n = (kt + 1) * BK;
#pragma unroll
            for (int p = 0; p < 2; ++p) {
                int chunk = p * 256 + tid;
                int r  = chunk >> 2;
                int cc = (chunk & 3) * 8;
                async_copy16(Ag + (size_t)(m0 + r) * K + k0n + cc, &As[nxt][r][cc]);
                async_copy16(Bg + (size_t)(n0 + r) * K + k0n + cc, &Bs[nxt][r][cc]);
            }
        }

        // A fragments (ISA 7.12.2): lane<16 K-chunks {0..7,16..23},
        //                           lane>=16 K-chunks {8..15,24..31}.
        v16h afr[4];
#pragma unroll
        for (int mi = 0; mi < 4; ++mi) {
            int mr = wm + mi * 16 + l16;
            v8h alo = *(const v8h*)&As[cur][mr][khalf * 8];
            v8h ahi = *(const v8h*)&As[cur][mr][16 + khalf * 8];
            afr[mi] = __builtin_shufflevector(alo, ahi,
                      0,1,2,3,4,5,6,7,8,9,10,11,12,13,14,15);
        }
        // B fragments: lane<16 col n K {0..7,16..23}; lane>=16 K {8..15,24..31}.
        v16h bfr[2];
#pragma unroll
        for (int ni = 0; ni < 2; ++ni) {
            int nr = wn + ni * 16 + l16;
            v8h blo = *(const v8h*)&Bs[cur][nr][khalf * 8];
            v8h bhi = *(const v8h*)&Bs[cur][nr][16 + khalf * 8];
            bfr[ni] = __builtin_shufflevector(blo, bhi,
                      0,1,2,3,4,5,6,7,8,9,10,11,12,13,14,15);
        }
#pragma unroll
        for (int mi = 0; mi < 4; ++mi)
#pragma unroll
            for (int ni = 0; ni < 2; ++ni)
                acc[mi][ni] = __builtin_amdgcn_wmma_f32_16x16x32_f16(
                    false, afr[mi], false, bfr[ni],
                    (short)0, acc[mi][ni], false, false);
        __syncthreads();                    // done reading `cur`
    }

    // ---- store D: VGPR r -> row = r + 8*(lane>=16), col = lane&15 ----
#pragma unroll
    for (int mi = 0; mi < 4; ++mi) {
#pragma unroll
        for (int ni = 0; ni < 2; ++ni) {
#pragma unroll
            for (int r = 0; r < 8; ++r) {
                int row = m0 + wm + mi * 16 + r + khalf * 8;
                int col = n0 + wn + ni * 16 + l16;
                float v = acc[mi][ni][r];
                if (mode == 0) {
                    ((float*)Cm)[(size_t)row * N + col] = v;
                } else if (mode == 1) {
                    int b = row >> 11, l = row & 2047;     // L = 2048
                    ((f16*)Cm)[((size_t)b * 512 + col) * 2048 + l] = (f16)v;
                } else {
                    ((f16*)Cm)[(size_t)row * N + col] = (f16)v;
                }
            }
        }
    }
}

// ------------------------- small helper kernels ----------------------------

__global__ __launch_bounds__(256)
void cvt_f32_to_f16(const float* __restrict__ in, f16* __restrict__ out, int n)
{
    int i = blockIdx.x * 256 + threadIdx.x;
    if (i < n) out[i] = (f16)in[i];
}

// Wt[n,k] = (f16) W[k,n]   (512x512)
__global__ __launch_bounds__(256)
void cvt_transpose_f16(const float* __restrict__ in, f16* __restrict__ out)
{
    int idx = blockIdx.x * 256 + threadIdx.x;
    if (idx >= DM * DM) return;
    int k = idx >> 9, n = idx & 511;
    out[(size_t)n * DM + k] = (f16)in[idx];
}

// FwT[row, t], row=2f(+1): cos(2*pi*f*t/L) / -sin(...); zero for f > 1024.
__global__ __launch_bounds__(256)
void fill_fwT(f16* __restrict__ FwT)
{
    int idx = blockIdx.x * 256 + threadIdx.x;
    if (idx >= NF2 * LL) return;
    int row = idx / LL, t = idx & (LL - 1);
    int f = row >> 1;
    float v = 0.f;
    if (f <= 1024) {
        float ang = 6.28318530717958647692f * (float)((f * t) & (LL - 1)) * (1.0f / LL);
        v = (row & 1) ? -sinf(ang) : cosf(ang);
    }
    FwT[idx] = (f16)v;
}

// BwT[tau, 2f(+1)] = a_f*cos(2*pi*f*tau/L) / -a_f*sin(...); a_f=1 for f in {0,1024} else 2.
__global__ __launch_bounds__(256)
void fill_bwT(f16* __restrict__ BwT)
{
    int idx = blockIdx.x * 256 + threadIdx.x;
    if (idx >= (size_t)LL * NF2) return;
    int tau = idx / NF2, r2 = idx % NF2;
    int f = r2 >> 1;
    float v = 0.f;
    if (f <= 1024) {
        float a = (f == 0 || f == 1024) ? 1.f : 2.f;
        float ang = 6.28318530717958647692f * (float)((f * tau) & (LL - 1)) * (1.0f / LL);
        v = (r2 & 1) ? -a * sinf(ang) : a * cosf(ang);
    }
    BwT[idx] = (f16)v;
}

// P = QF * conj(KF) / L   (f32 spectra -> f16 P)
__global__ __launch_bounds__(256)
void spectrum_mul(const float* __restrict__ QF, const float* __restrict__ KF,
                  f16* __restrict__ P)
{
    int p = blockIdx.x * 256 + threadIdx.x;
    if (p >= NCH * (NF2 / 2)) return;
    int row = p / (NF2 / 2), fp = p % (NF2 / 2);
    size_t base = (size_t)row * NF2 + 2 * fp;
    float qr = QF[base], qi = QF[base + 1];
    float kr = KF[base], ki = KF[base + 1];
    const float s = 1.0f / (float)LL;
    P[base]     = (f16)((qr * kr + qi * ki) * s);
    P[base + 1] = (f16)((qi * kr - qr * ki) * s);
}

// Per-channel top-38 over 2048 delays + softmax of the selected values.
__global__ __launch_bounds__(256)
void topk_softmax(const float* __restrict__ corr,
                  float* __restrict__ wts, int* __restrict__ dls)
{
    __shared__ float buf[LL];
    __shared__ float rv[256];
    __shared__ int   ri[256];
    __shared__ float tv[TOPK];
    __shared__ int   tix[TOPK];

    int c = blockIdx.x, t = threadIdx.x;
    const float* row = corr + (size_t)c * LL;
    for (int i = t; i < LL; i += 256) buf[i] = row[i];
    __syncthreads();

    for (int k = 0; k < TOPK; ++k) {
        float bv = -3.402823e38f; int bi = 0;
        for (int i = t; i < LL; i += 256) {
            float v = buf[i];
            if (v > bv) { bv = v; bi = i; }
        }
        rv[t] = bv; ri[t] = bi;
        __syncthreads();
        for (int s = 128; s > 0; s >>= 1) {
            if (t < s && rv[t + s] > rv[t]) { rv[t] = rv[t + s]; ri[t] = ri[t + s]; }
            __syncthreads();
        }
        if (t == 0) { tv[k] = rv[0]; tix[k] = ri[0]; buf[ri[0]] = -3.402823e38f; }
        __syncthreads();
    }

    if (t == 0) {
        float m = tv[0], s = 0.f;
        for (int k = 0; k < TOPK; ++k) { float e = __expf(tv[k] - m); tv[k] = e; s += e; }
        float inv = 1.f / s;
        for (int k = 0; k < TOPK; ++k) {
            wts[(size_t)c * TOPK + k] = tv[k] * inv;
            dls[(size_t)c * TOPK + k] = tix[k];
        }
    }
}

// context[b, l, hd] = sum_k w[k,b,hd] * V[b, (l+delay[k,b,hd]) % L, hd]
__global__ __launch_bounds__(512)
void aggregate(const f16* __restrict__ Vn, const float* __restrict__ wts,
               const int* __restrict__ dls, f16* __restrict__ ctx)
{
    int b = blockIdx.y, l0 = blockIdx.x * 8;
    int hd = threadIdx.x;                   // 0..511
    size_t c = (size_t)b * DM + hd;
    const f16* Vb = Vn + (size_t)b * LL * DM;
    float acc[8] = {};
    for (int k = 0; k < TOPK; ++k) {
        float w  = wts[c * TOPK + k];
        int  dly = dls[c * TOPK + k];
#pragma unroll
        for (int i = 0; i < 8; ++i) {
            int idx = (l0 + i + dly) & (LL - 1);
            acc[i] += w * (float)Vb[(size_t)idx * DM + hd];
        }
    }
#pragma unroll
    for (int i = 0; i < 8; ++i)
        ctx[((size_t)b * LL + l0 + i) * DM + hd] = (f16)acc[i];
}

// out = LayerNorm(fc + resid) * gamma + beta, one block per row of 512.
__global__ __launch_bounds__(256)
void add_layernorm(const float* __restrict__ fc, const float* __restrict__ resid,
                   const float* __restrict__ gamma, const float* __restrict__ beta,
                   float* __restrict__ out)
{
    __shared__ float s1[256], s2[256];
    __shared__ float mu, rs;
    int rowi = blockIdx.x, t = threadIdx.x;
    const float* f = fc    + (size_t)rowi * DM;
    const float* r = resid + (size_t)rowi * DM;
    float x0 = f[t] + r[t];
    float x1 = f[t + 256] + r[t + 256];
    s1[t] = x0 + x1;
    s2[t] = x0 * x0 + x1 * x1;
    __syncthreads();
    for (int s = 128; s > 0; s >>= 1) {
        if (t < s) { s1[t] += s1[t + s]; s2[t] += s2[t + s]; }
        __syncthreads();
    }
    if (t == 0) {
        float m = s1[0] * (1.0f / DM);
        float var = s2[0] * (1.0f / DM) - m * m;
        mu = m; rs = rsqrtf(var + LN_EPS);
    }
    __syncthreads();
    float* o = out + (size_t)rowi * DM;
    o[t]       = (x0 - mu) * rs * gamma[t]       + beta[t];
    o[t + 256] = (x1 - mu) * rs * gamma[t + 256] + beta[t + 256];
}

// ------------------------------- driver ------------------------------------

extern "C" void kernel_launch(void* const* d_in, const int* in_sizes, int n_in,
                              void* d_out, int out_size, void* d_ws, size_t ws_size,
                              hipStream_t stream)
{
    const float* inQ   = (const float*)d_in[0];
    const float* inK   = (const float*)d_in[1];
    const float* inV   = (const float*)d_in[2];
    const float* W_Q   = (const float*)d_in[3];
    const float* W_K   = (const float*)d_in[4];
    const float* W_V   = (const float*)d_in[5];
    const float* W_fc  = (const float*)d_in[6];
    const float* gamma = (const float*)d_in[7];
    const float* beta  = (const float*)d_in[8];
    float* outp = (float*)d_out;

    char* ws = (char*)d_ws;
    size_t off = 0;
    auto carve = [&](size_t bytes) -> void* {
        void* p = ws + off;
        off = (off + bytes + 255) & ~(size_t)255;
        return p;
    };
    const int M = BB * LL;                                   // 16384
    f16*   Qh  = (f16*)carve((size_t)M * DM * 2);            // f16 inputs
    f16*   Kh  = (f16*)carve((size_t)M * DM * 2);
    f16*   Vh  = (f16*)carve((size_t)M * DM * 2);
    f16*   WQt = (f16*)carve((size_t)DM * DM * 2);           // transposed weights
    f16*   WKt = (f16*)carve((size_t)DM * DM * 2);
    f16*   WVt = (f16*)carve((size_t)DM * DM * 2);
    f16*   WFt = (f16*)carve((size_t)DM * DM * 2);
    f16*   Qc  = (f16*)carve((size_t)NCH * LL * 2);          // [ch,t]; reused as ctx
    f16*   Kc  = (f16*)carve((size_t)NCH * LL * 2);
    f16*   Vn  = (f16*)carve((size_t)M * DM * 2);            // [b,l,hd]
    f16*   FwT = (f16*)carve((size_t)NF2 * LL * 2);          // fwd DFT, [N,K]
    f16*   BwT = (f16*)carve((size_t)LL * NF2 * 2);          // inv DFT, [N,K]
    float* QF  = (float*)carve((size_t)NCH * NF2 * 4);       // reused as corr
    float* KF  = (float*)carve((size_t)NCH * NF2 * 4);       // reused as fc_out
    f16*   P   = (f16*)carve((size_t)NCH * NF2 * 2);
    float* wts = (float*)carve((size_t)NCH * TOPK * 4);
    int*   dls = (int*)carve((size_t)NCH * TOPK * 4);

    float* corr = QF;   // QF dead after spectrum_mul
    f16*   ctx  = Qc;   // Qc dead after forward DFT
    float* fco  = KF;   // KF dead after spectrum_mul

    const int nIn = M * DM;      // 8388608
    const int nW  = DM * DM;     // 262144

    // 1) dtype conversions
    cvt_f32_to_f16<<<(nIn + 255) / 256, 256, 0, stream>>>(inQ, Qh, nIn);
    cvt_f32_to_f16<<<(nIn + 255) / 256, 256, 0, stream>>>(inK, Kh, nIn);
    cvt_f32_to_f16<<<(nIn + 255) / 256, 256, 0, stream>>>(inV, Vh, nIn);
    cvt_transpose_f16<<<(nW + 255) / 256, 256, 0, stream>>>(W_Q,  WQt);
    cvt_transpose_f16<<<(nW + 255) / 256, 256, 0, stream>>>(W_K,  WKt);
    cvt_transpose_f16<<<(nW + 255) / 256, 256, 0, stream>>>(W_V,  WVt);
    cvt_transpose_f16<<<(nW + 255) / 256, 256, 0, stream>>>(W_fc, WFt);

    // 2) DFT matrices (pre-transposed [N,K])
    fill_fwT<<<((size_t)NF2 * LL + 255) / 256, 256, 0, stream>>>(FwT);
    fill_bwT<<<((size_t)LL * NF2 + 255) / 256, 256, 0, stream>>>(BwT);

    // 3) projections
    gemm_wmma_f16<<<dim3(DM / 128, M / 128), 256, 0, stream>>>(Qh, WQt, Qc, M, DM, DM, 1);
    gemm_wmma_f16<<<dim3(DM / 128, M / 128), 256, 0, stream>>>(Kh, WKt, Kc, M, DM, DM, 1);
    gemm_wmma_f16<<<dim3(DM / 128, M / 128), 256, 0, stream>>>(Vh, WVt, Vn, M, DM, DM, 2);

    // 4) forward DFT
    gemm_wmma_f16<<<dim3(NF2 / 128, NCH / 128), 256, 0, stream>>>(Qc, FwT, QF, NCH, NF2, LL, 0);
    gemm_wmma_f16<<<dim3(NF2 / 128, NCH / 128), 256, 0, stream>>>(Kc, FwT, KF, NCH, NF2, LL, 0);

    // 5) spectrum product
    spectrum_mul<<<(NCH * (NF2 / 2) + 255) / 256, 256, 0, stream>>>(QF, KF, P);

    // 6) inverse DFT -> corr
    gemm_wmma_f16<<<dim3(LL / 128, NCH / 128), 256, 0, stream>>>(P, BwT, corr, NCH, LL, NF2, 0);

    // 7) top-k + softmax
    topk_softmax<<<NCH, 256, 0, stream>>>(corr, wts, dls);

    // 8) delayed aggregation
    aggregate<<<dim3(LL / 8, BB), 512, 0, stream>>>(Vn, wts, dls, ctx);

    // 9) output projection
    gemm_wmma_f16<<<dim3(DM / 128, M / 128), 256, 0, stream>>>(ctx, WFt, fco, M, DM, DM, 0);

    // 10) residual + LayerNorm
    add_layernorm<<<M, 256, 0, stream>>>(fco, inQ, gamma, beta, outp);
}
